// MoESparseLayer_53412213293897
// MI455X (gfx1250) — matrix operand order repaired and checked by reading, not compiled, not run on hardware
//
#include <hip/hip_runtime.h>
#include <math.h>

#define HIDDEN   1024
#define NEXPERT  8
#define FDIM     4096
#define TOKENS   8192
#define MT       64      // tokens per FFN block
#define FC       64      // F-chunk width
#define XS_STRIDE 1032   // halves, padded
#define HS_STRIDE 72     // halves, padded

typedef __attribute__((ext_vector_type(16))) __bf16 v16bf;
typedef __attribute__((ext_vector_type(8)))  float  v8f;

union BFrag { v16bf v; uint4 q[2]; };

__device__ __forceinline__ unsigned short f2bf(float f) {
  unsigned int u = __float_as_uint(f);
  u += 0x7FFFu + ((u >> 16) & 1u);           // round-to-nearest-even
  return (unsigned short)(u >> 16);
}

__device__ __forceinline__ float gelu_exact(float x) {
  return 0.5f * x * (1.0f + erff(x * 0.70710678118654752440f));
}

// ---------- fp32 -> bf16 copy of x ----------
__global__ __launch_bounds__(256) void convert_x_kernel(
    const float* __restrict__ in, unsigned short* __restrict__ outp, int n4) {
  int i = blockIdx.x * 256 + threadIdx.x;
  if (i >= n4) return;
  float4 f = ((const float4*)in)[i];
  uint2 o;
  o.x = (unsigned)f2bf(f.x) | ((unsigned)f2bf(f.y) << 16);
  o.y = (unsigned)f2bf(f.z) | ((unsigned)f2bf(f.w) << 16);
  ((uint2*)outp)[i] = o;
}

// ---------- per-expert transpose fp32 [R][C] -> bf16 [C][R] ----------
__global__ __launch_bounds__(256) void transpose_bf16_kernel(
    const float* __restrict__ in, unsigned short* __restrict__ outp, int R, int C) {
  __shared__ float tile[64][65];
  const size_t eoff = (size_t)blockIdx.z * R * C;
  in += eoff; outp += eoff;
  const int cb = blockIdx.x * 64;
  const int rb = blockIdx.y * 64;
  const int tx = threadIdx.x & 63;
  const int ty = threadIdx.x >> 6;
#pragma unroll
  for (int i = 0; i < 16; ++i) {
    int r = ty + i * 4;
    tile[r][tx] = in[(size_t)(rb + r) * C + (cb + tx)];
  }
  __syncthreads();
#pragma unroll
  for (int i = 0; i < 16; ++i) {
    int c = ty + i * 4;
    outp[(size_t)(cb + c) * R + (rb + tx)] = f2bf(tile[tx][c]);
  }
}

// ---------- gating: one wave32 per token, softmax + top-2 + scatter ----------
__global__ __launch_bounds__(256) void gating_kernel(
    const float* __restrict__ x, const float* __restrict__ gw,
    const float* __restrict__ gb, float* __restrict__ tkscore,
    int* __restrict__ counts, int* __restrict__ toklist) {
  const int wid  = threadIdx.x >> 5;
  const int lane = threadIdx.x & 31;
  const int t    = blockIdx.x * 8 + wid;
  const float* xr = x + (size_t)t * HIDDEN;
  float a[8] = {0.f, 0.f, 0.f, 0.f, 0.f, 0.f, 0.f, 0.f};
  for (int d = lane; d < HIDDEN; d += 32) {
    float xv = xr[d];
    const float4* g = (const float4*)(gw + d * 8);
    float4 g0 = g[0], g1 = g[1];
    a[0] += xv * g0.x; a[1] += xv * g0.y; a[2] += xv * g0.z; a[3] += xv * g0.w;
    a[4] += xv * g1.x; a[5] += xv * g1.y; a[6] += xv * g1.z; a[7] += xv * g1.w;
  }
#pragma unroll
  for (int e2 = 0; e2 < 8; ++e2)
    for (int off = 16; off > 0; off >>= 1)
      a[e2] += __shfl_xor(a[e2], off, 32);
  if (lane == 0) {
    float m = -3.0e38f;
#pragma unroll
    for (int e2 = 0; e2 < 8; ++e2) { a[e2] += gb[e2]; m = fmaxf(m, a[e2]); }
    float s = 0.f, p[8];
#pragma unroll
    for (int e2 = 0; e2 < 8; ++e2) { p[e2] = expf(a[e2] - m); s += p[e2]; }
    float inv = 1.f / s;
#pragma unroll
    for (int e2 = 0; e2 < 8; ++e2) p[e2] *= inv;
    int i1 = 0;
#pragma unroll
    for (int e2 = 1; e2 < 8; ++e2) if (p[e2] > p[i1]) i1 = e2;
    int i2 = (i1 == 0) ? 1 : 0;
#pragma unroll
    for (int e2 = 0; e2 < 8; ++e2)
      if (e2 != i1 && p[e2] > p[i2]) i2 = e2;
    int sel[2] = {i1, i2};
#pragma unroll
    for (int k = 0; k < 2; ++k) {
      int a_id = t * 2 + k;
      int ex = sel[k];
      tkscore[a_id] = p[ex];
      int pos = atomicAdd(&counts[ex], 1);
      toklist[ex * TOKENS + pos] = a_id;   // packed (token<<1 | k)
    }
  }
}

// ---------- fused grouped FFN: out += score * (gelu(X W1 + b1) W2 + b2) ----------
__global__ __launch_bounds__(256, 1) void moe_ffn_kernel(
    const unsigned short* __restrict__ xb,     // [TOKENS][HIDDEN] bf16
    const unsigned short* __restrict__ w1t,    // [E][FDIM][HIDDEN] bf16
    const unsigned short* __restrict__ w2t,    // [E][HIDDEN][FDIM] bf16
    const float* __restrict__ b1,              // [E][FDIM]
    const float* __restrict__ b2,              // [E][HIDDEN]
    const int* __restrict__ counts,
    const int* __restrict__ toklist,           // [E][TOKENS]
    const float* __restrict__ tkscore,         // [TOKENS*2]
    float* __restrict__ outp) {                // [TOKENS][HIDDEN]
  const int e = blockIdx.y;
  const int count = counts[e];
  const int base = blockIdx.x * MT;
  if (base >= count) return;

  extern __shared__ char smem[];
  unsigned short (*xs)[XS_STRIDE] = (unsigned short (*)[XS_STRIDE])smem;
  unsigned short (*hs)[HS_STRIDE] =
      (unsigned short (*)[HS_STRIDE])(smem + MT * XS_STRIDE * 2);
  int*   row_tok = (int*)(smem + MT * XS_STRIDE * 2 + MT * HS_STRIDE * 2);
  float* row_sc  = (float*)(row_tok + MT);

  const int tid  = threadIdx.x;
  const int lane = tid & 31;
  const int wid  = tid >> 5;
  const int hl   = lane & 15;   // low-16 lane id (row for A, col for B/C)
  const int hh   = lane >> 4;   // half-wave select

  if (tid < MT) {
    int tok = 0; float sc = 0.f;
    int r = base + tid;
    if (r < count) {
      int a = toklist[e * TOKENS + r];
      tok = a >> 1;
      sc  = tkscore[a];
    }
    row_tok[tid] = tok;
    row_sc[tid]  = sc;
  }
  __syncthreads();

  { // stage 64 token rows (bf16) into LDS
    int row = tid >> 2, q = tid & 3;
    const uint4* src = (const uint4*)(xb + (size_t)row_tok[row] * HIDDEN + q * 256);
    uint4* dst = (uint4*)(&xs[row][q * 256]);
#pragma unroll
    for (int i = 0; i < 32; ++i) dst[i] = src[i];
  }

  v8f acc[4][8] = {};   // persistent [64 x 128] fp32 output tile per wave

  const unsigned short* w1e = w1t + (size_t)e * FDIM * HIDDEN;
  const unsigned short* w2e = w2t + (size_t)e * HIDDEN * FDIM;
  const float* b1e = b1 + (size_t)e * FDIM;
  const int rt1 = wid >> 1;          // GEMM1: wave's H row-tile
  const int ct0 = (wid & 1) * 2;     // GEMM1: wave's first H col-tile
  const int nb2 = wid * 128;         // GEMM2: wave's output column base

  __syncthreads();

  for (int fc = 0; fc < FDIM; fc += FC) {
    if (fc + FC < FDIM) {
      __builtin_prefetch(w1e + (size_t)(fc + FC + ct0 * 16 + hl) * HIDDEN, 0, 1);
      __builtin_prefetch(w2e + (size_t)(nb2 + hl) * FDIM + fc + FC, 0, 1);
    }
    // ---- GEMM1: H[64][64] = gelu(X @ W1[:, fc:fc+64] + b1) ----
    v8f h0 = {}, h1 = {};
    const unsigned short* xrow  = &xs[rt1 * 16 + hl][0];
    const unsigned short* bcol0 = w1e + (size_t)(fc + ct0 * 16 + hl) * HIDDEN;
#pragma unroll 4
    for (int kk = 0; kk < HIDDEN; kk += 32) {
      BFrag A, B0, B1;
      A.q[0] = *(const uint4*)(xrow + kk + hh * 16);       // K = kk+hh*16 .. +7
      A.q[1] = *(const uint4*)(xrow + kk + hh * 16 + 8);   // K = kk+hh*16+8 ..
      const unsigned short* bp0 = bcol0 + kk + hh * 16;    // 16 contiguous K
      const unsigned short* bp1 = bp0 + 16 * HIDDEN;       // next col-tile
      B0.q[0] = *(const uint4*)bp0;  B0.q[1] = *(const uint4*)(bp0 + 8);
      B1.q[0] = *(const uint4*)bp1;  B1.q[1] = *(const uint4*)(bp1 + 8);
      h0 = __builtin_amdgcn_wmma_f32_16x16x32_bf16(false, A.v, false, B0.v,
                                                   (short)0, h0, false, false);
      h1 = __builtin_amdgcn_wmma_f32_16x16x32_bf16(false, A.v, false, B1.v,
                                                   (short)0, h1, false, false);
    }
    {
      float b1v0 = b1e[fc + ct0 * 16 + hl];
      float b1v1 = b1e[fc + (ct0 + 1) * 16 + hl];
      int rowb = rt1 * 16 + hh * 8;   // C/D layout: row = v + 8*hh
#pragma unroll
      for (int v = 0; v < 8; ++v) {
        hs[rowb + v][ct0 * 16 + hl]       = f2bf(gelu_exact(h0[v] + b1v0));
        hs[rowb + v][(ct0 + 1) * 16 + hl] = f2bf(gelu_exact(h1[v] + b1v1));
      }
    }
    __syncthreads();

    // ---- GEMM2: acc += H @ W2[fc:fc+64, wave's 128 cols] ----
#pragma unroll
    for (int ks = 0; ks < 2; ++ks) {
      BFrag B[8];
#pragma unroll
      for (int ct = 0; ct < 8; ++ct) {
        const unsigned short* bp =
            w2e + (size_t)(nb2 + ct * 16 + hl) * FDIM + fc + ks * 32 + hh * 16;
        B[ct].q[0] = *(const uint4*)bp;
        B[ct].q[1] = *(const uint4*)(bp + 8);
      }
#pragma unroll
      for (int rt = 0; rt < 4; ++rt) {
        BFrag A;
        const unsigned short* ap = &hs[rt * 16 + hl][ks * 32 + hh * 16];
        A.q[0] = *(const uint4*)ap;
        A.q[1] = *(const uint4*)(ap + 8);
#pragma unroll
        for (int ct = 0; ct < 8; ++ct)
          acc[rt][ct] = __builtin_amdgcn_wmma_f32_16x16x32_bf16(
              false, A.v, false, B[ct].v, (short)0, acc[rt][ct], false, false);
      }
    }
    __syncthreads();
  }

  // ---- epilogue: out[token] += score * (acc + b2) ----
#pragma unroll
  for (int ct = 0; ct < 8; ++ct) {
    int col = nb2 + ct * 16 + hl;
    float b2v = b2[(size_t)e * HIDDEN + col];
#pragma unroll
    for (int rt = 0; rt < 4; ++rt) {
#pragma unroll
      for (int v = 0; v < 8; ++v) {
        int row = rt * 16 + hh * 8 + v;
        float val = (acc[rt][ct][v] + b2v) * row_sc[row];  // score=0 pads => +0
        atomicAdd(&outp[(size_t)row_tok[row] * HIDDEN + col], val);
      }
    }
  }
}

extern "C" void kernel_launch(void* const* d_in, const int* in_sizes, int n_in,
                              void* d_out, int out_size, void* d_ws, size_t ws_size,
                              hipStream_t stream) {
  const float* x  = (const float*)d_in[0];
  const float* gw = (const float*)d_in[1];
  const float* gb = (const float*)d_in[2];
  const float* w1 = (const float*)d_in[3];
  const float* b1 = (const float*)d_in[4];
  const float* w2 = (const float*)d_in[5];
  const float* b2 = (const float*)d_in[6];
  float* outp = (float*)d_out;

  // workspace layout
  char* ws = (char*)d_ws;
  int*   counts  = (int*)ws;                    // 256 B (8 ints + pad)
  float* tkscore = (float*)(ws + 256);          // 64 KB
  int*   toklist = (int*)(ws + 256 + 65536);    // 256 KB (8 x 8192 ints)
  unsigned short* xb  = (unsigned short*)(ws + 327936);          // 16 MB
  unsigned short* w1t = xb + (size_t)TOKENS * HIDDEN;            // 64 MB
  unsigned short* w2t = w1t + (size_t)NEXPERT * FDIM * HIDDEN;   // 64 MB

  hipMemsetAsync(d_out, 0, (size_t)out_size * sizeof(float), stream);  // also aux_loss
  hipMemsetAsync(ws, 0, 256, stream);                                  // expert counts

  convert_x_kernel<<<(TOKENS * HIDDEN / 4 + 255) / 256, 256, 0, stream>>>(
      x, xb, TOKENS * HIDDEN / 4);
  transpose_bf16_kernel<<<dim3(FDIM / 64, HIDDEN / 64, NEXPERT), 256, 0, stream>>>(
      w1, w1t, HIDDEN, FDIM);   // w1t[e][f][d]
  transpose_bf16_kernel<<<dim3(HIDDEN / 64, FDIM / 64, NEXPERT), 256, 0, stream>>>(
      w2, w2t, FDIM, HIDDEN);   // w2t[e][d][f]
  gating_kernel<<<TOKENS / 8, 256, 0, stream>>>(x, gw, gb, tkscore, counts, toklist);

  size_t smem = (size_t)MT * XS_STRIDE * 2 + (size_t)MT * HS_STRIDE * 2 + (size_t)MT * 8;
  moe_ffn_kernel<<<dim3(TOKENS / MT, NEXPERT), 256, smem, stream>>>(
      xb, w1t, w2t, b1, b2, counts, toklist, tkscore, outp);

  (void)in_sizes; (void)n_in; (void)ws_size;
}
